// LinOSSLayer_53429393163009
// MI455X (gfx1250) — matrix operand order repaired and checked by reading, not compiled, not run on hardware
//
#include <hip/hip_runtime.h>
#include <stddef.h>
#include <stdint.h>

// ---------------------------------------------------------------------------
// LinOSS layer for MI455X (gfx1250, wave32, WMMA, async-to-LDS).
// Dims: Hd=256, P=64, BSZ=16, L=2048.  N2 = 2P = 128.
// ---------------------------------------------------------------------------

typedef __attribute__((ext_vector_type(2)))  float  v2f;
typedef __attribute__((ext_vector_type(8)))  float  v8f;
typedef __attribute__((ext_vector_type(16))) __bf16 v16bf;

union AB16 { v16bf v; unsigned int u[8]; };

__device__ __forceinline__ unsigned int pack_bf16(float lo, float hi) {
    union { __bf16 b; unsigned short s; } cl, ch;
    cl.b = (__bf16)lo; ch.b = (__bf16)hi;
    return (unsigned int)cl.s | ((unsigned int)ch.s << 16);
}

// Async copy of 16 contiguous bytes: global (saddr + voff) -> LDS (ldsaddr).
// GLOBAL_LOAD_ASYNC_TO_LDS_B128, tracked with ASYNCcnt (ISA 15.18.3 op 98).
__device__ __forceinline__ void async_b128(unsigned ldsaddr, unsigned voff,
                                           const void* saddr) {
    asm volatile("global_load_async_to_lds_b128 %0, %1, %2"
                 :: "v"(ldsaddr), "v"(voff), "s"(saddr) : "memory");
}
__device__ __forceinline__ void wait_async0() {
    asm volatile("s_wait_asynccnt 0x0" ::: "memory");
}
__device__ __forceinline__ unsigned lds_addr32(const void* p) {
    return (unsigned)(uintptr_t)p;   // LDS aperture: addr[31:0] = LDS offset
}

// ---------------------------------------------------------------------------
// K1: operand repack.
//   Bmat  f32 [128 n][256 k]  (transposed, K contiguous, for f32 WMMA in K2)
//   Cp    u32 [64 pair][256 h], pair packs (C[h][p][0], -C[h][p][1]) as bf16
//   W1p/W2p u32 [128 pair][256 n], pair packs (W[n][2k], W[n][2k+1]) as bf16
// ---------------------------------------------------------------------------
__global__ __launch_bounds__(256) void k1_prep(
    const float* __restrict__ Bw, const float* __restrict__ Cw,
    const float* __restrict__ W1, const float* __restrict__ W2,
    float* __restrict__ Bmat, unsigned int* __restrict__ Cp,
    unsigned int* __restrict__ W1p, unsigned int* __restrict__ W2p)
{
    const int tid = blockIdx.x * blockDim.x + threadIdx.x;
    const int np  = gridDim.x * blockDim.x;
    for (int idx = tid; idx < 128 * 256; idx += np) {  // Bmat[n][k=h]
        int n = idx >> 8, h = idx & 255;
        int p = n >> 1,  r = n & 1;
        Bmat[idx] = Bw[(p * 256 + h) * 2 + r];         // B is (P, Hd, 2)
    }
    for (int idx = tid; idx < 64 * 256; idx += np) {   // Cp[pair=p][h]
        int p = idx >> 8, h = idx & 255;
        float lo =  Cw[(h * 64 + p) * 2 + 0];          // C is (Hd, P, 2)
        float hi = -Cw[(h * 64 + p) * 2 + 1];
        Cp[idx] = pack_bf16(lo, hi);
    }
    for (int idx = tid; idx < 128 * 256; idx += np) {  // W?p[pair][n]
        int pi = idx >> 8, n = idx & 255;
        W1p[idx] = pack_bf16(W1[n * 256 + 2 * pi], W1[n * 256 + 2 * pi + 1]);
        W2p[idx] = pack_bf16(W2[n * 256 + 2 * pi], W2[n * 256 + 2 * pi + 1]);
    }
}

// ---------------------------------------------------------------------------
// K2: Bu GEMM, fp32 WMMA 16x16x4.  M=(b,l)=32768, N=128, K=256.
// Block = 256 thr (8 waves); block tile 128(l) x 128(n); wave tile 16 x 128.
// B staged in LDS [128 n][260 k] f32 (133 KB) via async B128 copies; each
// B-fragment is then one aligned ds_load_b64 straight into a register pair.
// u streamed + global_prefetch.  Out: Bu f32 (B, L, 128) raw.
// ---------------------------------------------------------------------------
__global__ __launch_bounds__(256) void k2_bu(
    const float* __restrict__ u, const float* __restrict__ Bmat,
    float* __restrict__ Bu)
{
    extern __shared__ float bsm[];                 // [128][260]
    const int t  = threadIdx.x;
    const int b  = blockIdx.x >> 4;
    const int l0 = (blockIdx.x & 15) * 128;

    // async-stage B operand: 8192 x 16B chunks, 32 per thread
    for (int c = t; c < 128 * 64; c += 256) {
        const int n  = c >> 6;                     // 64 chunks per n-row
        const int kq = (c & 63) * 4;
        async_b128(lds_addr32(&bsm[n * 260 + kq]), (unsigned)(c * 16), Bmat);
    }
    wait_async0();
    __syncthreads();

    const int wave = t >> 5, lane = t & 31;
    const int g = lane >> 4, ml = lane & 15;
    const int lrow = l0 + wave * 16 + ml;          // A-frag row for this lane
    const float* ub = u + (size_t)b * 256 * 2048;

    const v8f vz = {0.f,0.f,0.f,0.f,0.f,0.f,0.f,0.f};
    v8f acc[8];
#pragma unroll
    for (int nt = 0; nt < 8; ++nt) acc[nt] = vz;

    for (int k0 = 0; k0 < 256; k0 += 4) {
        // A frag: lanes 0-15 hold K=k0,k0+1 ; lanes 16-31 hold K=k0+2,k0+3
        const int kk = k0 + 2 * g;
        v2f a;
        a[0] = ub[(size_t)(kk + 0) * 2048 + lrow];
        a[1] = ub[(size_t)(kk + 1) * 2048 + lrow];
        if (k0 + 8 < 256)
            __builtin_prefetch(&ub[(size_t)(k0 + 8 + 2 * g) * 2048 + lrow], 0, 1);
#pragma unroll
        for (int nt = 0; nt < 8; ++nt) {
            const int n = nt * 16 + ml;
            // B frag VGPR j: K = k0 + j + 2g  -> contiguous pair in [n][k]
            const v2f bf = *(const v2f*)&bsm[n * 260 + k0 + 2 * g];
            acc[nt] = __builtin_amdgcn_wmma_f32_16x16x4_f32(
                false, a, false, bf, (short)0, acc[nt], false, false);
        }
    }

    float* ob = Bu + (size_t)b * 2048 * 128;
#pragma unroll
    for (int nt = 0; nt < 8; ++nt) {
        const int n = nt * 16 + ml;
#pragma unroll
        for (int i = 0; i < 8; ++i) {
            const int l = l0 + wave * 16 + i + 8 * g;   // C/D: VGPR i -> M=i / i+8
            ob[(size_t)l * 128 + n] = acc[nt][i];
        }
    }
}

// ---------------------------------------------------------------------------
// K3: chunked parallel scan of  [z;x]_l = M [z;x]_{l-1} + [c1;c2]*Bu_l.
// Block = (b, group of 8 p).  t -> (nl = t&15 over 8p x 2r, chunk c = t>>4).
// 16 chunks x 128 steps; Bu slice (128 KB) async-staged into LDS;
// Hillis-Steele over chunk carries, level matrices = squarings of M^128.
// ---------------------------------------------------------------------------
__global__ __launch_bounds__(256) void k3_scan(
    const float* __restrict__ Bu, const float* __restrict__ A_diag,
    const float* __restrict__ steps, float* __restrict__ xout)
{
    extern __shared__ float sm3[];
    float* bsh = sm3;                 // [2048][16]
    float* zs  = sm3 + 2048 * 16;     // [256]
    float* xs  = zs + 256;            // [256]

    const int t  = threadIdx.x;
    const int b  = blockIdx.x >> 3;
    const int pg = blockIdx.x & 7;
    const int n0 = pg * 16;
    const int nl = t & 15;
    const int c  = t >> 4;            // chunk 0..15
    const int p  = pg * 8 + (nl >> 1);

    const float Ar  = fmaxf(A_diag[p], 0.f);
    const float dt  = 1.f / (1.f + __expf(-steps[p]));
    const float S   = 1.f / (1.f + dt * dt * Ar);
    const float m11 = 1.f - dt * dt * Ar * S;
    const float m12 = -dt * Ar * S;
    const float m21 = dt * S;
    const float m22 = S;
    const float c1  = m11 * dt, c2 = m21 * dt;

    // async-stage this block's Bu slice: per l, 16 floats = four 16B chunks
    const float* bub = Bu + (size_t)b * 2048 * 128 + n0;
    for (int ch = t; ch < 2048 * 4; ch += 256) {
        const int l = ch >> 2, q = (ch & 3) * 4;
        async_b128(lds_addr32(&bsh[l * 16 + q]),
                   (unsigned)((l * 128 + q) * 4), bub);
    }
    wait_async0();
    __syncthreads();

    // local carry over this chunk's 128 steps (from zero state)
    const int lb = c * 128;
    float z = 0.f, x = 0.f;
    for (int i = 0; i < 128; ++i) {
        float v  = bsh[(lb + i) * 16 + nl];
        float zn = m11 * z + m12 * x + c1 * v;
        float xn = m21 * z + m22 * x + c2 * v;
        z = zn; x = xn;
    }
    // M^128 by repeated squaring
    float a00 = m11, a01 = m12, a10 = m21, a11 = m22;
    for (int s = 0; s < 7; ++s) {
        float n00 = a00*a00 + a01*a10, n01 = a00*a01 + a01*a11;
        float n10 = a10*a00 + a11*a10, n11 = a10*a01 + a11*a11;
        a00 = n00; a01 = n01; a10 = n10; a11 = n11;
    }
    zs[t] = z; xs[t] = x;
    // inclusive scan over chunks; combine(earlier,self): F = Mlvl*F_e + F_s
    for (int d = 1; d < 16; d <<= 1) {
        __syncthreads();
        float pz = 0.f, px = 0.f;
        if (c >= d) { pz = zs[t - d * 16]; px = xs[t - d * 16]; }
        __syncthreads();
        if (c >= d) {
            z = a00 * pz + a01 * px + z;
            x = a10 * pz + a11 * px + x;
            zs[t] = z; xs[t] = x;
        }
        float n00 = a00*a00 + a01*a10, n01 = a00*a01 + a01*a11;
        float n10 = a10*a00 + a11*a10, n11 = a10*a01 + a11*a11;
        a00 = n00; a01 = n01; a10 = n10; a11 = n11;
    }
    __syncthreads();
    float zp = 0.f, xp = 0.f;
    if (c > 0) { zp = zs[t - 16]; xp = xs[t - 16]; }   // exclusive prefix

    float* xob = xout + (size_t)b * 2048 * 128 + n0;
    for (int i = 0; i < 128; ++i) {
        float v  = bsh[(lb + i) * 16 + nl];
        float zn = m11 * zp + m12 * xp + c1 * v;
        float xn = m21 * zp + m22 * xp + c2 * v;
        xob[(size_t)(lb + i) * 128 + nl] = xn;
        zp = zn; xp = xn;
    }
}

// ---------------------------------------------------------------------------
// K4: fused  ys = x @ Ccat + D*u ; g = gelu(ys) ;
//            out = (g@W1^T + b1) * sigmoid(g@W2^T + b2) ; transpose store.
// Block = 256 thr (8 waves); tile 128(l) x 256(h); bf16 WMMA 16x16x32.
// LDS: xt u32[128][69] (bf16-pair A tile), ut f32[128][257] (u tile, reused
// as result tile for the coalesced transposed store), gt bf16[128][266].
// ---------------------------------------------------------------------------
__global__ __launch_bounds__(256) void k4_out(
    const float* __restrict__ xout, const float* __restrict__ u,
    const unsigned int* __restrict__ Cp, const unsigned int* __restrict__ W1p,
    const unsigned int* __restrict__ W2p, const float* __restrict__ Dv,
    const float* __restrict__ b1, const float* __restrict__ b2,
    float* __restrict__ out)
{
    extern __shared__ char smem[];
    unsigned int*  xt = (unsigned int*)smem;                         // [128][69]
    float*         ut = (float*)(smem + 128 * 69 * 4);               // [128][257]
    unsigned short* gt = (unsigned short*)(smem + 128*69*4 + 128*257*4); // [128][266]
    const unsigned int* gt32 = (const unsigned int*)gt;              // stride 133
    float* rt = ut;                                                  // reuse

    const int t  = threadIdx.x;
    const int b  = blockIdx.x >> 4;
    const int l0 = (blockIdx.x & 15) * 128;

    // stage x tile as packed bf16 pairs (pair = (r0, r1) of one p)
    {
        const float* xb = xout + ((size_t)b * 2048 + l0) * 128;
        for (int idx = t; idx < 128 * 64; idx += 256) {
            int pr = idx & 63, l = idx >> 6;
            xt[l * 69 + pr] = pack_bf16(xb[(size_t)l * 128 + pr * 2],
                                        xb[(size_t)l * 128 + pr * 2 + 1]);
        }
    }
    // stage u tile transposed: ut[l][h]
    {
        const float* row = u + ((size_t)b * 256 + t) * 2048 + l0;   // h = t
        for (int v = 0; v < 32; ++v) {
            float4 d = *(const float4*)(row + v * 4);
            int lb = v * 4;
            ut[(lb + 0) * 257 + t] = d.x;
            ut[(lb + 1) * 257 + t] = d.y;
            ut[(lb + 2) * 257 + t] = d.z;
            ut[(lb + 3) * 257 + t] = d.w;
        }
    }
    __syncthreads();

    const int wave = t >> 5, lane = t & 31;
    const int g = lane >> 4, ml = lane & 15;
    const int wl = wave * 16;
    const int arow = wl + ml;                       // A-frag row (local l)
    const v8f vz = {0.f,0.f,0.f,0.f,0.f,0.f,0.f,0.f};

    // ---- C projection: (16 l) x (256 h), K = 128 -----------------------
    v8f acc[16];
#pragma unroll
    for (int nt = 0; nt < 16; ++nt) acc[nt] = vz;
#pragma unroll
    for (int ks = 0; ks < 4; ++ks) {
        AB16 a;
#pragma unroll
        for (int v = 0; v < 8; ++v) {
            int pi = ks * 16 + (v < 4 ? v + 4 * g : 8 + (v - 4) + 4 * g);
            a.u[v] = xt[arow * 69 + pi];
        }
#pragma unroll
        for (int nt = 0; nt < 16; ++nt) {
            const int n = nt * 16 + ml;
            AB16 bb;
#pragma unroll
            for (int v = 0; v < 8; ++v)
                bb.u[v] = Cp[(ks * 16 + (v < 4 ? v + 4 * g : 8 + (v - 4) + 4 * g)) * 256 + n];
            acc[nt] = __builtin_amdgcn_wmma_f32_16x16x32_bf16(
                false, a.v, false, bb.v, (short)0, acc[nt], false, false);
        }
    }
    // ---- epilogue: + D*u, exact GELU, store bf16 g tile ----------------
#pragma unroll
    for (int nt = 0; nt < 16; ++nt) {
        const int h = nt * 16 + ml;
        const float dv = Dv[h];
#pragma unroll
        for (int i = 0; i < 8; ++i) {
            const int lr = wl + i + 8 * g;
            float v  = acc[nt][i] + dv * ut[lr * 257 + h];
            float ge = 0.5f * v * (1.f + erff(v * 0.70710678118654752f));
            union { __bf16 bx; unsigned short s; } cv; cv.bx = (__bf16)ge;
            gt[lr * 266 + h] = cv.s;
        }
    }
    __syncthreads();

    // ---- gated MLP: two GEMMs (16 l) x (256 h), K = 256 ----------------
    for (int half = 0; half < 2; ++half) {
        v8f a1[8], a2[8];
#pragma unroll
        for (int j = 0; j < 8; ++j) { a1[j] = vz; a2[j] = vz; }
#pragma unroll
        for (int ks = 0; ks < 8; ++ks) {
            AB16 a;
#pragma unroll
            for (int v = 0; v < 8; ++v) {
                int pi = ks * 16 + (v < 4 ? v + 4 * g : 8 + (v - 4) + 4 * g);
                a.u[v] = gt32[arow * 133 + pi];
            }
#pragma unroll
            for (int j = 0; j < 8; ++j) {
                const int n = (half * 8 + j) * 16 + ml;
                AB16 w1f, w2f;
#pragma unroll
                for (int v = 0; v < 8; ++v) {
                    const int pi = ks * 16 + (v < 4 ? v + 4 * g : 8 + (v - 4) + 4 * g);
                    w1f.u[v] = W1p[pi * 256 + n];
                    w2f.u[v] = W2p[pi * 256 + n];
                }
                a1[j] = __builtin_amdgcn_wmma_f32_16x16x32_bf16(
                    false, a.v, false, w1f.v, (short)0, a1[j], false, false);
                a2[j] = __builtin_amdgcn_wmma_f32_16x16x32_bf16(
                    false, a.v, false, w2f.v, (short)0, a2[j], false, false);
            }
        }
#pragma unroll
        for (int j = 0; j < 8; ++j) {
            const int h = (half * 8 + j) * 16 + ml;
            const float bb1 = b1[h], bb2 = b2[h];
#pragma unroll
            for (int i = 0; i < 8; ++i) {
                const int lr = wl + i + 8 * g;
                float y1 = a1[j][i] + bb1;
                float y2 = a2[j][i] + bb2;
                rt[lr * 257 + h] = y1 * (1.f / (1.f + __expf(-y2)));
            }
        }
    }
    __syncthreads();

    // ---- coalesced transposed store: out[b][h][l0+l] -------------------
    float* outb = out + (size_t)b * 256 * 2048 + l0;
    for (int pass = 0; pass < 32; ++pass) {
        const int h  = pass * 8 + (t >> 5);
        const int li = (t & 31) * 4;
        float4 d;
        d.x = rt[(li + 0) * 257 + h];
        d.y = rt[(li + 1) * 257 + h];
        d.z = rt[(li + 2) * 257 + h];
        d.w = rt[(li + 3) * 257 + h];
        *(float4*)(outb + (size_t)h * 2048 + li) = d;
    }
}

// ---------------------------------------------------------------------------
// Host launcher. Inputs: u, A_diag, B, C, D, steps, W1, b1, W2, b2 (all f32).
// ---------------------------------------------------------------------------
extern "C" void kernel_launch(void* const* d_in, const int* in_sizes, int n_in,
                              void* d_out, int out_size, void* d_ws, size_t ws_size,
                              hipStream_t stream) {
    const float* u      = (const float*)d_in[0];
    const float* A_diag = (const float*)d_in[1];
    const float* Bw     = (const float*)d_in[2];
    const float* Cw     = (const float*)d_in[3];
    const float* Dv     = (const float*)d_in[4];
    const float* steps  = (const float*)d_in[5];
    const float* W1     = (const float*)d_in[6];
    const float* b1     = (const float*)d_in[7];
    const float* W2     = (const float*)d_in[8];
    const float* b2     = (const float*)d_in[9];
    float* out = (float*)d_out;

    char* ws = (char*)d_ws;
    float*        Bmat = (float*)(ws + 0);                 // 131072 B
    unsigned int* Cp   = (unsigned int*)(ws + 131072);     //  65536 B
    unsigned int* W1p  = (unsigned int*)(ws + 196608);     // 131072 B
    unsigned int* W2p  = (unsigned int*)(ws + 327680);     // 131072 B
    float*        Bu   = (float*)(ws + 458752);            // 16 MB
    float*        xo   = (float*)(ws + 458752 + 16777216); // 16 MB

    k1_prep<<<dim3(256), dim3(256), 0, stream>>>(Bw, Cw, W1, W2, Bmat, Cp, W1p, W2p);

    const size_t smem2 = 128 * 260 * 4;                     // 133120
    k2_bu<<<dim3(256), dim3(256), smem2, stream>>>(u, Bmat, Bu);

    const size_t smem3 = 2048 * 16 * 4 + 512 * 4;           // 133120
    k3_scan<<<dim3(128), dim3(256), smem3, stream>>>(Bu, A_diag, steps, xo);

    const size_t smem4 = 128*69*4 + 128*257*4 + 128*266*2;  // 235008
    k4_out<<<dim3(256), dim3(256), smem4, stream>>>(xo, u, Cp, W1p, W2p, Dv, b1, b2, out);
}
